// GatedDeltaState_21122649162392
// MI455X (gfx1250) — matrix operand order repaired and checked by reading, not compiled, not run on hardware
//
#include <hip/hip_runtime.h>
#include <hip/hip_bf16.h>
#include <math.h>

// ---------------------------------------------------------------------------
// Gated DeltaNet for MI455X (gfx1250, wave32, WMMA, TDM)
//  K1  proj_gemm : X(32768x512) @ [Wk|Wv|Wq|Wb|Wa|Wg]^T  -> bf16 bufs (WMMA bf16)
//  K1b knorm     : L2-normalize k per (b,t,h) over D=32   (shfl reduce)
//  K2  scan      : chunked WY delta-rule, 64 blocks x 1 wave,
//                  chunk staging via TDM tensor_load_to_lds, 6x 32^3 WMMA/chunk
//  K3a norm_gate : rmsnorm * silu(gate)                   (shfl reduce)
//  K3b out_gemm  : ctx(32768x256) @ W_out^T -> fp32 out   (WMMA bf16)
// ---------------------------------------------------------------------------

typedef __attribute__((ext_vector_type(16))) __bf16    v16bf;
typedef __attribute__((ext_vector_type(8)))  float     v8f;
typedef __attribute__((ext_vector_type(4)))  unsigned  v4u;
typedef __attribute__((ext_vector_type(8)))  unsigned  v8u;

#define HIDDEN 512
#define SDIM   256
#define NH     8
#define HD     32
#define BS     8
#define TT     4096
#define NROW   (BS*TT)          /* 32768 rows of (b,t) */

__device__ __forceinline__ int a_kmap(int e, int lane) {
    // 16-bit A matrix 16x32: lane L holds M=L&15; element e -> K
    return e + (((lane >> 4) & 1) << 4);
}
__device__ __forceinline__ int b_kmap(int e, int lane) {
    // 16-bit B matrix 32x16: lane L holds N=L&15; element e -> K
    return e + (((lane >> 4) & 1) << 4);
}
__device__ __forceinline__ v8f zero8() {
    v8f z = {0.f,0.f,0.f,0.f,0.f,0.f,0.f,0.f};
    return z;
}
__device__ __forceinline__ float sigmoidf_(float x) {
    return __builtin_amdgcn_rcpf(1.f + __expf(-x));
}

// ---------------------------------------------------------------------------
// TDM: async-DMA a 2D tile (32 rows x 32 bf16, row stride 256 elems) from
// global memory into LDS.  D# per CDNA5 ISA ch.8: group0 = {count, lds_addr,
// global_addr, type=2}; group1 = {data_size=2B, tensor dims (large, no OOB
// clamp), tile 32x32, dim0_stride=256}; groups 2/3 zero (2D tensor).
// Completion tracked with TENSORcnt.
// ---------------------------------------------------------------------------
__device__ __forceinline__ void tdm_load_tile_32x32_bf16(unsigned lds_addr,
                                                         const __bf16* gptr)
{
    unsigned long long ga = (unsigned long long)(uintptr_t)gptr;
    v4u g0;
    g0[0] = 1u;                                           // count=1, user mode
    g0[1] = lds_addr;                                     // LDS byte address
    g0[2] = (unsigned)(ga & 0xffffffffull);               // global_addr[31:0]
    g0[3] = (unsigned)((ga >> 32) & 0x01ffffffull)        // global_addr[56:32]
          | (2u << 30);                                   // type = 2 (image)
    v8u g1;
    g1[0] = 0x00010000u;            // workgroup_mask=0, data_size=1 (2 bytes)
    g1[1] = 0xffff0000u;            // tensor_dim0 lo16 (dim0 = 0x7fffffff)
    g1[2] = 0xffff7fffu;            // tensor_dim0 hi | tensor_dim1 lo16
    g1[3] = 0x00207fffu;            // tensor_dim1 hi | tile_dim0 = 32
    g1[4] = 32u;                    // tile_dim1 = 32, tile_dim2 = 0
    g1[5] = 256u;                   // tensor_dim0_stride = 256 (lo32)
    g1[6] = 0u;                     // stride hi | tensor_dim1_stride lo
    g1[7] = 0u;
    v4u gz = {0u, 0u, 0u, 0u};
    asm volatile("tensor_load_to_lds %0, %1, %2, %3"
                 :: "s"(g0), "s"(g1), "s"(gz), "s"(gz)
                 : "memory");
}

// ---------------------------------------------------------------------------
// K1: fused projection GEMM.  M=32768, N=1536 (6 x 256 concat), K=512.
// Block tile 128x128, 8 waves (4x2), wave tile 32x64 = 2x4 WMMA tiles.
// ---------------------------------------------------------------------------
__global__ __launch_bounds__(256) void gdn_proj_gemm(
    const float* __restrict__ X,
    const float* __restrict__ Wk, const float* __restrict__ Wv,
    const float* __restrict__ Wq, const float* __restrict__ Wb,
    const float* __restrict__ Wa, const float* __restrict__ Wg,
    const float* __restrict__ bb, const float* __restrict__ ba,
    __bf16* __restrict__ outK, __bf16* __restrict__ outV,
    __bf16* __restrict__ outQ, __bf16* __restrict__ outB,
    __bf16* __restrict__ outA, __bf16* __restrict__ outG)
{
    __shared__ alignas(32) __bf16 ldsA[8][32][16];   // 8 m-tiles, frag layout
    __shared__ alignas(32) __bf16 ldsB[8][32][16];   // 8 n-tiles, frag layout

    const int tid  = threadIdx.x;
    const int lane = tid & 31;
    const int wave = tid >> 5;
    const int wm   = wave >> 1;          // 0..3
    const int wn   = wave & 1;           // 0..1
    const int m0   = blockIdx.x * 128;
    const int n0   = blockIdx.y * 128;
    const int p    = n0 >> 8;            // uniform per block (128 | 256)

    const float* Wsel = (p==0)?Wk : (p==1)?Wv : (p==2)?Wq : (p==3)?Wb : (p==4)?Wa : Wg;
    __bf16* Osel      = (p==0)?outK : (p==1)?outV : (p==2)?outQ : (p==3)?outB : (p==4)?outA : outG;

    v8f acc[2][4];
    #pragma unroll
    for (int i=0;i<2;i++)
        #pragma unroll
        for (int j=0;j<4;j++) acc[i][j] = zero8();

    for (int k0 = 0; k0 < HIDDEN; k0 += 32) {
        __builtin_prefetch((const void*)&X[(m0 + (tid>>1))*HIDDEN + k0 + 32], 0, 1);
        // stage A tile (128x32 fp32 -> bf16, fragment order)
        #pragma unroll
        for (int i=0;i<16;i++) {
            int idx = tid + i*256;          // 0..4095
            int mL  = idx >> 5, kk = idx & 31;
            float v = X[(m0 + mL)*HIDDEN + k0 + kk];
            int L = (mL & 15) + ((kk & 16) ? 16 : 0);
            int e = kk & 15;
            ldsA[mL >> 4][L][e] = (__bf16)v;
        }
        // stage B tile (weights, 128x32 fp32 -> bf16, fragment order)
        #pragma unroll
        for (int i=0;i<16;i++) {
            int idx = tid + i*256;
            int nL  = idx >> 5, kk = idx & 31;
            int c   = (n0 + nL) & 255;
            float v = Wsel[c*HIDDEN + k0 + kk];
            int L = (nL & 15) + ((kk & 16) ? 16 : 0);
            int e = kk & 15;
            ldsB[nL >> 4][L][e] = (__bf16)v;
        }
        __syncthreads();
        v16bf af[2], bf[4];
        #pragma unroll
        for (int i=0;i<2;i++) af[i] = *(const v16bf*)&ldsA[wm*2 + i][lane][0];
        #pragma unroll
        for (int j=0;j<4;j++) bf[j] = *(const v16bf*)&ldsB[wn*4 + j][lane][0];
        #pragma unroll
        for (int i=0;i<2;i++)
            #pragma unroll
            for (int j=0;j<4;j++)
                acc[i][j] = __builtin_amdgcn_wmma_f32_16x16x32_bf16(
                    false, af[i], false, bf[j], (short)0, acc[i][j], false, false);
        __syncthreads();
    }

    // epilogue: sigmoid(+bias) for beta/alpha, store bf16
    const int mhalf = (lane >> 4) * 8;
    const int n16   = lane & 15;
    #pragma unroll
    for (int i=0;i<2;i++) {
        #pragma unroll
        for (int j=0;j<4;j++) {
            int c = ((n0 + (wn*4 + j)*16 + n16) & 255);
            #pragma unroll
            for (int r=0;r<8;r++) {
                int row = m0 + (wm*2 + i)*16 + r + mhalf;
                float v = acc[i][j][r];
                if (p == 3)      v = sigmoidf_(v + bb[c]);
                else if (p == 4) v = sigmoidf_(v + ba[c]);
                Osel[(size_t)row*SDIM + c] = (__bf16)v;
            }
        }
    }
}

// ---------------------------------------------------------------------------
// K1b: k <- k / max(||k||_2, 1e-12) over D=32 per (b,t,h). One wave per row.
// ---------------------------------------------------------------------------
__global__ __launch_bounds__(256) void gdn_knorm(__bf16* __restrict__ k)
{
    int row  = blockIdx.x*8 + (threadIdx.x >> 5);   // over NROW*NH
    int lane = threadIdx.x & 31;
    size_t idx = (size_t)row*HD + lane;
    float v = (float)k[idx];
    float s = v*v;
    #pragma unroll
    for (int m=16;m>0;m>>=1) s += __shfl_xor(s, m, 32);
    float n = fmaxf(sqrtf(s), 1e-12f);
    k[idx] = (__bf16)(v / n);
}

// ---------------------------------------------------------------------------
// K2: chunked WY delta-rule scan. One block = one wave = one (b,h).
// Per chunk (C=32): stage K,V,Q,beta,alpha via TDM; G=KK^T, GQ=QK^T,
// R0k=K S0^T, R0q=Q S0^T (WMMA); triangular solve for W (registers);
// O=(tril GQ)W, dS=W^T K (WMMA).
// ---------------------------------------------------------------------------
template<class FA, class FB, class FC>
__device__ __forceinline__ void gemm32(int lane, FA fa, FB fb, FC fc)
{
    const int m16 = lane & 15;
    #pragma unroll
    for (int mt=0; mt<2; ++mt) {
        #pragma unroll
        for (int nt=0; nt<2; ++nt) {
            v16bf a, b;
            #pragma unroll
            for (int e=0;e<16;e++) {
                a[e] = (__bf16)fa(mt*16 + m16, a_kmap(e, lane));
                b[e] = (__bf16)fb(nt*16 + m16, b_kmap(e, lane));
            }
            v8f c = zero8();
            c = __builtin_amdgcn_wmma_f32_16x16x32_bf16(
                    false, a, false, b, (short)0, c, false, false);
            #pragma unroll
            for (int r=0;r<8;r++)
                fc(mt*16 + r + ((lane >> 4) << 3), nt*16 + m16, c[r]);
        }
    }
}

__global__ __launch_bounds__(32) void gdn_scan(
    const __bf16* __restrict__ kb, const __bf16* __restrict__ vb,
    const __bf16* __restrict__ qb, const __bf16* __restrict__ beb,
    const __bf16* __restrict__ alb,
    const float* __restrict__ state_in,
    float* __restrict__ ctx, float* __restrict__ state_out)
{
    // raw bf16 chunk tiles (TDM destinations)
    __shared__ alignas(16) __bf16 Kr[32][32], Vr[32][32], Qr[32][32];
    __shared__ alignas(16) __bf16 Br[32][32], Ar[32][32];
    // fp32 working set
    __shared__ float S0[32][32];
    __shared__ float CA[32][32];                 // cumulative alpha product
    __shared__ float G [32][32], GQ[32][32];
    __shared__ float R0k[32][32], R0q[32][32];
    __shared__ float Wl[32][32];

    const int bh = blockIdx.x, b = bh >> 3, h = bh & 7;
    const int lane = threadIdx.x;

    const unsigned ldsK = (unsigned)(uintptr_t)&Kr[0][0];
    const unsigned ldsV = (unsigned)(uintptr_t)&Vr[0][0];
    const unsigned ldsQ = (unsigned)(uintptr_t)&Qr[0][0];
    const unsigned ldsB = (unsigned)(uintptr_t)&Br[0][0];
    const unsigned ldsA = (unsigned)(uintptr_t)&Ar[0][0];

    for (int i=0;i<32;i++)
        S0[i][lane] = state_in[(size_t)bh*1024 + i*32 + lane];

    #pragma clang loop unroll(disable)
    for (int chunk = 0; chunk < TT/32; ++chunk) {
        const int t0 = chunk*32;
        const size_t base = (size_t)(b*TT + t0)*SDIM + h*HD;   // tile origin

        // async DMA the five 32x32 bf16 tiles into LDS (Tensor Data Mover)
        tdm_load_tile_32x32_bf16(ldsK, kb  + base);
        tdm_load_tile_32x32_bf16(ldsV, vb  + base);
        tdm_load_tile_32x32_bf16(ldsQ, qb  + base);
        tdm_load_tile_32x32_bf16(ldsB, beb + base);
        tdm_load_tile_32x32_bf16(ldsA, alb + base);
        __builtin_amdgcn_s_wait_tensorcnt(0);

        float cp = 1.f;
        for (int t=0;t<32;t++) { cp *= (float)Ar[t][lane]; CA[t][lane] = cp; }

        // Gram / cross matrices + state projections (bf16 WMMA)
        gemm32(lane, [&](int t,int d){ return Kr[t][d]; },
                     [&](int s,int d){ return Kr[s][d]; },
                     [&](int t,int s,float v){ G[t][s]  = v; });
        gemm32(lane, [&](int t,int d){ return Qr[t][d]; },
                     [&](int s,int d){ return Kr[s][d]; },
                     [&](int t,int s,float v){ GQ[t][s] = v; });
        gemm32(lane, [&](int t,int j){ return Kr[t][j]; },
                     [&](int i,int j){ return S0[i][j]; },
                     [&](int t,int i,float v){ R0k[t][i] = v; });
        gemm32(lane, [&](int t,int j){ return Qr[t][j]; },
                     [&](int i,int j){ return S0[i][j]; },
                     [&](int t,int i,float v){ R0q[t][i] = v; });

        // triangular solve for w_t (lane owns column d; w kept in VGPRs)
        float w[32];
        #pragma unroll
        for (int t=0;t<32;t++) {
            float acc = 0.f;
            #pragma unroll
            for (int s=0;s<t;s++) acc += w[s] * G[t][s];
            float bt = (float)Br[t][lane];
            float wt = bt*(float)Vr[t][lane]*__builtin_amdgcn_rcpf(CA[t][lane])
                     - bt*R0k[t][lane] - bt*acc;
            w[t] = wt;
            Wl[t][lane] = wt;
        }

        // readout: out[t][d] = CA[t][d]*(R0q[t][d] + sum_{s<=t} GQ[t][s]*W[s][d])
        gemm32(lane, [&](int t,int s){ return (s <= t) ? GQ[t][s] : 0.f; },
                     [&](int d,int s){ return Wl[s][d]; },
                     [&](int t,int d,float v){
                         float o = CA[t][d]*(R0q[t][d] + v);
                         ctx[(size_t)(b*TT + t0 + t)*SDIM + h*HD + d] = o;
                     });
        // state update: S0[i][j] = CA[31][i]*(S0[i][j] + (W^T K)[i][j])
        gemm32(lane, [&](int i,int s){ return Wl[s][i]; },
                     [&](int j,int s){ return Kr[s][j]; },
                     [&](int i,int j,float v){
                         S0[i][j] = CA[31][i]*(S0[i][j] + v);
                     });
    }

    for (int i=0;i<32;i++)
        state_out[(size_t)bh*1024 + i*32 + lane] = S0[i][lane];
}

// ---------------------------------------------------------------------------
// K3a: rmsnorm(ctx)*norm_w * silu(gate) -> bf16.  One wave per (b,t) row.
// ---------------------------------------------------------------------------
__global__ __launch_bounds__(256) void gdn_norm_gate(
    const float* __restrict__ ctx, const float* __restrict__ norm_w,
    const __bf16* __restrict__ gate, __bf16* __restrict__ ctxn)
{
    int row  = blockIdx.x*8 + (threadIdx.x >> 5);
    int lane = threadIdx.x & 31;
    float x[8]; float ss = 0.f;
    #pragma unroll
    for (int j=0;j<8;j++) {
        x[j] = ctx[(size_t)row*SDIM + j*32 + lane];
        ss += x[j]*x[j];
    }
    #pragma unroll
    for (int m=16;m>0;m>>=1) ss += __shfl_xor(ss, m, 32);
    float r = rsqrtf(ss * (1.f/256.f) + 1e-6f);
    #pragma unroll
    for (int j=0;j<8;j++) {
        int c = j*32 + lane;
        float g = (float)gate[(size_t)row*SDIM + c];
        ctxn[(size_t)row*SDIM + c] = (__bf16)(x[j]*r*norm_w[c] * g * sigmoidf_(g));
    }
}

// ---------------------------------------------------------------------------
// K3b: out = ctxn(32768x256) @ W_out^T(256x512) -> fp32.  WMMA bf16.
// ---------------------------------------------------------------------------
__global__ __launch_bounds__(256) void gdn_out_gemm(
    const __bf16* __restrict__ A, const float* __restrict__ Wout,
    float* __restrict__ out)
{
    __shared__ alignas(32) __bf16 ldsA[8][32][16];
    __shared__ alignas(32) __bf16 ldsB[8][32][16];
    const int tid = threadIdx.x, lane = tid & 31, wave = tid >> 5;
    const int wm = wave >> 1, wn = wave & 1;
    const int m0 = blockIdx.x * 128;
    const int n0 = blockIdx.y * 128;

    v8f acc[2][4];
    #pragma unroll
    for (int i=0;i<2;i++)
        #pragma unroll
        for (int j=0;j<4;j++) acc[i][j] = zero8();

    for (int k0 = 0; k0 < SDIM; k0 += 32) {
        #pragma unroll
        for (int i=0;i<16;i++) {
            int idx = tid + i*256;
            int mL  = idx >> 5, kk = idx & 31;
            int L = (mL & 15) + ((kk & 16) ? 16 : 0);
            int e = kk & 15;
            ldsA[mL >> 4][L][e] = A[(size_t)(m0 + mL)*SDIM + k0 + kk];
        }
        #pragma unroll
        for (int i=0;i<16;i++) {
            int idx = tid + i*256;
            int nL  = idx >> 5, kk = idx & 31;
            float v = Wout[(size_t)(n0 + nL)*SDIM + k0 + kk];
            int L = (nL & 15) + ((kk & 16) ? 16 : 0);
            int e = kk & 15;
            ldsB[nL >> 4][L][e] = (__bf16)v;
        }
        __syncthreads();
        v16bf af[2], bf[4];
        #pragma unroll
        for (int i=0;i<2;i++) af[i] = *(const v16bf*)&ldsA[wm*2 + i][lane][0];
        #pragma unroll
        for (int j=0;j<4;j++) bf[j] = *(const v16bf*)&ldsB[wn*4 + j][lane][0];
        #pragma unroll
        for (int i=0;i<2;i++)
            #pragma unroll
            for (int j=0;j<4;j++)
                acc[i][j] = __builtin_amdgcn_wmma_f32_16x16x32_bf16(
                    false, af[i], false, bf[j], (short)0, acc[i][j], false, false);
        __syncthreads();
    }

    const int mhalf = (lane >> 4) * 8;
    const int n16   = lane & 15;
    #pragma unroll
    for (int i=0;i<2;i++)
        #pragma unroll
        for (int j=0;j<4;j++) {
            int n = n0 + (wn*4 + j)*16 + n16;
            #pragma unroll
            for (int r=0;r<8;r++) {
                int row = m0 + (wm*2 + i)*16 + r + mhalf;
                out[(size_t)row*HIDDEN + n] = acc[i][j][r];
            }
        }
}

// ---------------------------------------------------------------------------
extern "C" void kernel_launch(void* const* d_in, const int* in_sizes, int n_in,
                              void* d_out, int out_size, void* d_ws, size_t ws_size,
                              hipStream_t stream)
{
    const float* X     = (const float*)d_in[0];
    const float* state = (const float*)d_in[1];
    const float* Wk    = (const float*)d_in[2];
    const float* Wv    = (const float*)d_in[3];
    const float* Wq    = (const float*)d_in[4];
    const float* Wb    = (const float*)d_in[5];
    const float* bb    = (const float*)d_in[6];
    const float* Wa    = (const float*)d_in[7];
    const float* ba    = (const float*)d_in[8];
    const float* Wout  = (const float*)d_in[9];
    const float* Wg    = (const float*)d_in[10];
    const float* normw = (const float*)d_in[11];

    char* ws = (char*)d_ws;
    const size_t PE = (size_t)NROW * SDIM;            // 8,388,608 elements
    __bf16* kB   = (__bf16*)(ws);
    __bf16* vB   = kB + PE;
    __bf16* qB   = vB + PE;
    __bf16* beB  = qB + PE;
    __bf16* alB  = beB + PE;
    __bf16* gB   = alB + PE;
    float*  ctx  = (float*)(ws + 6*PE*sizeof(__bf16));
    __bf16* ctxn = (__bf16*)(ws + 6*PE*sizeof(__bf16) + PE*sizeof(float));

    float* out      = (float*)d_out;
    float* stateOut = out + (size_t)NROW * HIDDEN;

    gdn_proj_gemm<<<dim3(NROW/128, (6*SDIM)/128), 256, 0, stream>>>(
        X, Wk, Wv, Wq, Wb, Wa, Wg, bb, ba, kB, vB, qB, beB, alB, gB);
    gdn_knorm<<<dim3((NROW*NH)/8), 256, 0, stream>>>(kB);
    gdn_scan<<<dim3(BS*NH), 32, 0, stream>>>(
        kB, vB, qB, beB, alB, state, ctx, stateOut);
    gdn_norm_gate<<<dim3(NROW/8), 256, 0, stream>>>(ctx, normw, gB, ctxn);
    gdn_out_gemm<<<dim3(NROW/128, HIDDEN/128), 256, 0, stream>>>(ctxn, Wout, out);
}
